// QuantumWalkRetriever_34308198760632
// MI455X (gfx1250) — matrix operand order repaired and checked by reading, not compiled, not run on hardware
//
#include <hip/hip_runtime.h>
#include <hip/hip_bf16.h>

// ---------------------------------------------------------------------------
// QuantumWalkRetriever for MI455X (gfx1250, wave32, WMMA)
//
// coin MLP (768->128->8) and path MLP (800 padded ->128->1) run through
// v_wmma_f32_16x16x32_f16 with M=32 per wave (two strips): each B fragment
// load feeds two WMMAs, halving L2 weight traffic vs M=16.
// Walk steps are linear => normalization telescopes to one final norm.
// ---------------------------------------------------------------------------

typedef __attribute__((ext_vector_type(16))) _Float16 v16h;
typedef __attribute__((ext_vector_type(8)))  _Float16 v8h;
typedef __attribute__((ext_vector_type(8)))  float    v8f;

#define B_   16
#define N_   8192
#define D_   384
#define KN_  16
#define K_   8
#define H_   128
#define BN_  (B_ * N_)

union HalfPack { v16h v; v8h h[2]; };

static __device__ __forceinline__ v16h make_frag(v8h lo, v8h hi) {
  HalfPack u; u.h[0] = lo; u.h[1] = hi; return u.v;
}

static __device__ __forceinline__ void cvt16(_Float16* dst, const float4* src) {
#pragma unroll
  for (int t = 0; t < 4; ++t) {
    float4 f = src[t];
    dst[t * 4 + 0] = (_Float16)f.x;
    dst[t * 4 + 1] = (_Float16)f.y;
    dst[t * 4 + 2] = (_Float16)f.z;
    dst[t * 4 + 3] = (_Float16)f.w;
  }
}

static __device__ __forceinline__ void fma4(float4& o, const float4& s, const float4& a) {
  o.x += s.x * a.x; o.y += s.y * a.y; o.z += s.z * a.z; o.w += s.w * a.w;
}

// ---------------------------------------------------------------------------
// Pack f32 weights into WMMA B-fragment layout (f16), tiles of 32(K) x 16(N).
// Within a tile: lane L holds 16 contiguous f16 at offset L*16.
//   k = kc*32 + (L<16 ? 0 : 8) + (e<8 ? e : e+8),  col = nt*16 + (L&15)
// mode 0: coin_w1 [768x128]           (24 kc x 8 nt tiles)
// mode 1: coin_w2 [128x8] pad to x16  ( 4 kc x 1 nt tiles)
// mode 2: path_w1 [776x128] K pad 800 (rows: [0,392)=src, [392,416)=0,
//                                      [416,800)=src row-24; 25 kc x 8 nt)
// ---------------------------------------------------------------------------
__global__ __launch_bounds__(256) void pack_weights_kernel(
    const float* __restrict__ w, _Float16* __restrict__ out, int mode, int total) {
  int id = blockIdx.x * 256 + threadIdx.x;
  if (id >= total) return;
  int e = id & 15;
  int L = (id >> 4) & 31;
  int t = id >> 9;
  int kc, nt;
  if (mode == 1) { kc = t; nt = 0; } else { kc = t >> 3; nt = t & 7; }
  int kp  = kc * 32 + ((L < 16) ? 0 : 8) + ((e < 8) ? e : e + 8);
  int col = nt * 16 + (L & 15);
  float val = 0.f;
  if (mode == 0) {
    val = w[kp * H_ + col];
  } else if (mode == 1) {
    if (col < K_) val = w[kp * K_ + col];
  } else {
    if (kp < 392)       val = w[kp * H_ + col];
    else if (kp >= 416) val = w[(kp - 24) * H_ + col];
  }
  out[id] = (_Float16)val;
}

// ---------------------------------------------------------------------------
// Coin MLP: amps[b,n,:] = relu(concat(sent, q) @ W1 + b1) @ W2 + b2
// Block = 256 rows, 8 waves; wave w owns rows w*32..w*32+31 (two 16-row
// strips, wave-private LDS => no block barriers anywhere).
// ---------------------------------------------------------------------------
__global__ __launch_bounds__(256) void coin_mlp_kernel(
    const float* __restrict__ sent, const float* __restrict__ qe,
    const _Float16* __restrict__ w1p, const float* __restrict__ b1,
    const _Float16* __restrict__ w2p, const float* __restrict__ b2,
    float* __restrict__ amps) {
  __shared__ _Float16 ldsA[8][32][48];   // per-wave A chunk: 32 rows x 32 (pad 48)
  __shared__ _Float16 ldsH[8][16][136];  // per-wave hidden scratch (one strip)

  const int tid   = threadIdx.x;
  const int wave  = tid >> 5, lane = tid & 31;
  const int mbase = blockIdx.x * 256;
  const int b     = mbase >> 13;          // N_ = 8192
  const int nbase = mbase & (N_ - 1);
  const float* sentB = sent + ((size_t)b * N_ + nbase) * D_;
  const float* qB    = qe + (size_t)b * D_;

  v8f acc0[8] = {};
  v8f acc1[8] = {};

  // staging: lane stages one full 32-wide row (row = lane within wave tile)
  _Float16* sdst = &ldsA[wave][lane][0];
  const size_t rowOff = (size_t)(wave * 32 + lane) * D_;

  const int ar   = lane & 15;
  const int koff = (lane < 16) ? 0 : 8;

  for (int kc = 0; kc < 24; ++kc) {
    const int k0 = kc * 32;               // 32-run never straddles 384
    if (k0 < D_) {
      cvt16(sdst,      (const float4*)(sentB + rowOff + k0));
      cvt16(sdst + 16, (const float4*)(sentB + rowOff + k0 + 16));
    } else {
      cvt16(sdst,      (const float4*)(qB + (k0 - D_)));
      cvt16(sdst + 16, (const float4*)(qB + (k0 - D_ + 16)));
    }

    v16h a0 = make_frag(*(const v8h*)&ldsA[wave][ar][koff],
                        *(const v8h*)&ldsA[wave][ar][koff + 16]);
    v16h a1 = make_frag(*(const v8h*)&ldsA[wave][16 + ar][koff],
                        *(const v8h*)&ldsA[wave][16 + ar][koff + 16]);

    const _Float16* wt = w1p + (size_t)kc * (8 * 512) + lane * 16;
#pragma unroll
    for (int nt = 0; nt < 8; ++nt) {
      v8h blo = *(const v8h*)(wt + nt * 512);
      v8h bhi = *(const v8h*)(wt + nt * 512 + 8);
      v16h bf = make_frag(blo, bhi);
      acc0[nt] = __builtin_amdgcn_wmma_f32_16x16x32_f16(
          false, a0, false, bf, (short)0, acc0[nt], false, false);
      acc1[nt] = __builtin_amdgcn_wmma_f32_16x16x32_f16(
          false, a1, false, bf, (short)0, acc1[nt], false, false);
    }
  }

  const int colL   = lane & 15;
  const int rowoff = (lane < 16) ? 0 : 8;

#pragma unroll
  for (int s = 0; s < 2; ++s) {
    const v8f* accS = s ? acc1 : acc0;
    // bias + ReLU -> wave-private hidden scratch (f16)
#pragma unroll
    for (int nt = 0; nt < 8; ++nt) {
      float bias = b1[nt * 16 + colL];
#pragma unroll
      for (int v = 0; v < 8; ++v) {
        float x = accS[nt][v] + bias;
        ldsH[wave][rowoff + v][nt * 16 + colL] = (_Float16)fmaxf(x, 0.f);
      }
    }
    // layer 2: hidden[16x128] @ W2[128x16(pad)] = 4 WMMAs
    v8f acc2 = {};
#pragma unroll
    for (int kc = 0; kc < 4; ++kc) {
      const int ko = kc * 32 + ((lane < 16) ? 0 : 8);
      v16h af = make_frag(*(const v8h*)&ldsH[wave][ar][ko],
                          *(const v8h*)&ldsH[wave][ar][ko + 16]);
      const _Float16* wt = w2p + kc * 512 + lane * 16;
      v16h bf = make_frag(*(const v8h*)(wt), *(const v8h*)(wt + 8));
      acc2 = __builtin_amdgcn_wmma_f32_16x16x32_f16(
          false, af, false, bf, (short)0, acc2, false, false);
    }
    if (colL < K_) {
      float bias2 = b2[colL];
#pragma unroll
      for (int v = 0; v < 8; ++v) {
        int node = nbase + wave * 32 + s * 16 + rowoff + v;
        amps[((size_t)b * N_ + node) * K_ + colL] = acc2[v] + bias2;
      }
    }
  }
}

// ---------------------------------------------------------------------------
// One (un-normalized) walk step: out[i] = s(i)*a(i) + sum_j s(nbr_j)*a(nbr_j)
// first=1: s(x) == 1/256 (uniform initial state); gathers hit L2 (8MB set).
// ---------------------------------------------------------------------------
__global__ __launch_bounds__(256) void walk_step_kernel(
    const float* __restrict__ st, const float* __restrict__ amps,
    const int* __restrict__ nbrs, float* __restrict__ out, int first) {
  int i = blockIdx.x * 256 + threadIdx.x;
  if (i >= BN_) return;
  int b = i >> 13;
  const float4* ap = (const float4*)amps;
  const float4* sp = (const float4*)st;
  const float c0 = 0.00390625f;  // 1/sqrt(N*K) = 1/256
  const float4 cv = make_float4(c0, c0, c0, c0);

  float4 a0 = ap[(size_t)i * 2], a1 = ap[(size_t)i * 2 + 1];
  float4 o0 = make_float4(0.f, 0.f, 0.f, 0.f), o1 = o0;
  if (first) { fma4(o0, cv, a0); fma4(o1, cv, a1); }
  else {
    fma4(o0, sp[(size_t)i * 2], a0);
    fma4(o1, sp[(size_t)i * 2 + 1], a1);
  }

  const int* nb = nbrs + (size_t)i * KN_;
#pragma unroll 4
  for (int j = 0; j < KN_; ++j) {
    int t = nb[j];
    if ((unsigned)t < (unsigned)N_) {
      size_t nix = ((size_t)(b << 13) + t) * 2;
      float4 na0 = ap[nix], na1 = ap[nix + 1];
      if (first) { fma4(o0, cv, na0); fma4(o1, cv, na1); }
      else       { fma4(o0, sp[nix], na0); fma4(o1, sp[nix + 1], na1); }
    }
  }
  float4* op = (float4*)out;
  op[(size_t)i * 2] = o0;
  op[(size_t)i * 2 + 1] = o1;
}

__global__ void zero_norm_kernel(float* __restrict__ normsq) {
  if (threadIdx.x < B_) normsq[threadIdx.x] = 0.f;
}

__global__ __launch_bounds__(256) void norm_kernel(
    const float* __restrict__ st, float* __restrict__ normsq) {
  __shared__ float red[256];
  int tid = threadIdx.x;
  int i = blockIdx.x * 256 + tid;         // 512 blocks, 32 per batch
  const float4* sp = (const float4*)st;
  float4 a = sp[(size_t)i * 2], c = sp[(size_t)i * 2 + 1];
  float s = a.x * a.x + a.y * a.y + a.z * a.z + a.w * a.w +
            c.x * c.x + c.y * c.y + c.z * c.z + c.w * c.w;
  red[tid] = s;
  __syncthreads();
  for (int off = 128; off > 0; off >>= 1) {
    if (tid < off) red[tid] += red[tid + off];
    __syncthreads();
  }
  if (tid == 0) atomicAdd(&normsq[blockIdx.x >> 5], red[0]);
}

// ---------------------------------------------------------------------------
// Path MLP: logits = relu([sent | state/||.|| pad | q] @ W1 + b1) @ w2 + b2
// K padded 776 -> 800 (25 chunks of 32). M=32 per wave; layer 2 (1 output
// col) done as an f32 dot over wave-private hidden scratch, one strip at a
// time with block barriers around the cross-wave dot phase.
// ---------------------------------------------------------------------------
__global__ __launch_bounds__(256) void path_mlp_kernel(
    const float* __restrict__ sent, const float* __restrict__ qe,
    const float* __restrict__ state, const float* __restrict__ normsq,
    const _Float16* __restrict__ w1p, const float* __restrict__ b1,
    const float* __restrict__ w2, const float* __restrict__ b2,
    float* __restrict__ logits) {
  __shared__ _Float16 ldsA[8][32][48];
  __shared__ _Float16 ldsH[8][16][136];

  const int tid   = threadIdx.x;
  const int wave  = tid >> 5, lane = tid & 31;
  const int mbase = blockIdx.x * 256;
  const int b     = mbase >> 13;
  const int nbase = mbase & (N_ - 1);
  const float* sentB = sent + ((size_t)b * N_ + nbase) * D_;
  const float* qB    = qe + (size_t)b * D_;
  const float nsq = normsq[b];
  const float inv = (nsq > 0.f) ? rsqrtf(nsq) : 1.f;

  v8f acc0[8] = {};
  v8f acc1[8] = {};

  _Float16* sdst = &ldsA[wave][lane][0];
  const int  rowG = wave * 32 + lane;     // block-local row this lane stages

  const int ar   = lane & 15;
  const int koff = (lane < 16) ? 0 : 8;

  for (int kc = 0; kc < 25; ++kc) {
    const int k0 = kc * 32;               // boundaries 384 / 416 are 32-aligned
    if (k0 < 384) {
      cvt16(sdst,      (const float4*)(sentB + (size_t)rowG * D_ + k0));
      cvt16(sdst + 16, (const float4*)(sentB + (size_t)rowG * D_ + k0 + 16));
    } else if (k0 == 384) {
      const float* stp = state + ((size_t)b * N_ + nbase + rowG) * K_;
#pragma unroll
      for (int e = 0; e < 8; ++e)  sdst[e] = (_Float16)(stp[e] * inv);
#pragma unroll
      for (int e = 8; e < 32; ++e) sdst[e] = (_Float16)0.f;
    } else {
      cvt16(sdst,      (const float4*)(qB + (k0 - 416)));
      cvt16(sdst + 16, (const float4*)(qB + (k0 - 416 + 16)));
    }

    v16h a0 = make_frag(*(const v8h*)&ldsA[wave][ar][koff],
                        *(const v8h*)&ldsA[wave][ar][koff + 16]);
    v16h a1 = make_frag(*(const v8h*)&ldsA[wave][16 + ar][koff],
                        *(const v8h*)&ldsA[wave][16 + ar][koff + 16]);

    const _Float16* wt = w1p + (size_t)kc * (8 * 512) + lane * 16;
#pragma unroll
    for (int nt = 0; nt < 8; ++nt) {
      v8h blo = *(const v8h*)(wt + nt * 512);
      v8h bhi = *(const v8h*)(wt + nt * 512 + 8);
      v16h bf = make_frag(blo, bhi);
      acc0[nt] = __builtin_amdgcn_wmma_f32_16x16x32_f16(
          false, a0, false, bf, (short)0, acc0[nt], false, false);
      acc1[nt] = __builtin_amdgcn_wmma_f32_16x16x32_f16(
          false, a1, false, bf, (short)0, acc1[nt], false, false);
    }
  }

  const int colL   = lane & 15;
  const int rowoff = (lane < 16) ? 0 : 8;

#pragma unroll
  for (int s = 0; s < 2; ++s) {
    const v8f* accS = s ? acc1 : acc0;
#pragma unroll
    for (int nt = 0; nt < 8; ++nt) {
      float bias = b1[nt * 16 + colL];
#pragma unroll
      for (int v = 0; v < 8; ++v) {
        float x = accS[nt][v] + bias;
        ldsH[wave][rowoff + v][nt * 16 + colL] = (_Float16)fmaxf(x, 0.f);
      }
    }
    __syncthreads();  // cross-wave dot reads every wave's scratch
    if (tid < 128) {
      const int wv = tid >> 4, r = tid & 15;
      float d = b2[0];
#pragma unroll 8
      for (int c = 0; c < H_; ++c) d += (float)ldsH[wv][r][c] * w2[c];
      logits[(size_t)b * N_ + nbase + wv * 32 + s * 16 + r] = d;
    }
    __syncthreads();  // protect scratch before strip 1 overwrites it
  }
}

// ---------------------------------------------------------------------------
extern "C" void kernel_launch(void* const* d_in, const int* in_sizes, int n_in,
                              void* d_out, int out_size, void* d_ws, size_t ws_size,
                              hipStream_t stream) {
  (void)in_sizes; (void)n_in; (void)out_size; (void)ws_size;

  const float* sent = (const float*)d_in[0];
  const float* qe   = (const float*)d_in[1];
  const int*   nbrs = (const int*)d_in[2];
  const float* cw1  = (const float*)d_in[3];
  const float* cb1  = (const float*)d_in[4];
  const float* cw2  = (const float*)d_in[5];
  const float* cb2  = (const float*)d_in[6];
  const float* pw1  = (const float*)d_in[7];
  const float* pb1  = (const float*)d_in[8];
  const float* pw2  = (const float*)d_in[9];
  const float* pb2  = (const float*)d_in[10];
  float* logits = (float*)d_out;

  // workspace carve-up (256B aligned)
  char* ws = (char*)d_ws;
  size_t off = 0;
  auto carve = [&](size_t bytes) {
    size_t o = off;
    off = (off + bytes + 255) & ~(size_t)255;
    return o;
  };
  const size_t cw1p_elems = 24 * 8 * 512;   // 98304 f16
  const size_t cw2p_elems = 4 * 512;        // 2048 f16
  const size_t pw1p_elems = 25 * 8 * 512;   // 102400 f16
  _Float16* cw1p = (_Float16*)(ws + carve(cw1p_elems * 2));
  _Float16* cw2p = (_Float16*)(ws + carve(cw2p_elems * 2));
  _Float16* pw1p = (_Float16*)(ws + carve(pw1p_elems * 2));
  float* amps   = (float*)(ws + carve((size_t)BN_ * K_ * 4));
  float* stateA = (float*)(ws + carve((size_t)BN_ * K_ * 4));
  float* stateB = (float*)(ws + carve((size_t)BN_ * K_ * 4));
  float* normsq = (float*)(ws + carve(B_ * 4));

  zero_norm_kernel<<<1, 32, 0, stream>>>(normsq);
  pack_weights_kernel<<<(int)(cw1p_elems / 256), 256, 0, stream>>>(cw1, cw1p, 0, (int)cw1p_elems);
  pack_weights_kernel<<<(int)(cw2p_elems / 256), 256, 0, stream>>>(cw2, cw2p, 1, (int)cw2p_elems);
  pack_weights_kernel<<<(int)(pw1p_elems / 256), 256, 0, stream>>>(pw1, pw1p, 2, (int)pw1p_elems);

  coin_mlp_kernel<<<BN_ / 256, 256, 0, stream>>>(sent, qe, cw1p, cb1, cw2p, cb2, amps);

  walk_step_kernel<<<BN_ / 256, 256, 0, stream>>>(amps /*unused*/, amps, nbrs, stateA, 1);
  walk_step_kernel<<<BN_ / 256, 256, 0, stream>>>(stateA, amps, nbrs, stateB, 0);
  walk_step_kernel<<<BN_ / 256, 256, 0, stream>>>(stateB, amps, nbrs, stateA, 0);

  norm_kernel<<<BN_ / 256, 256, 0, stream>>>(stateA, normsq);

  path_mlp_kernel<<<BN_ / 256, 256, 0, stream>>>(sent, qe, stateA, normsq,
                                                 pw1p, pb1, pw2, pb2, logits);
}
